// HGTAssetCorrelator_90220083019924
// MI455X (gfx1250) — matrix operand order repaired and checked by reading, compiled, not run on hardware
//
#include <hip/hip_runtime.h>
#include <hip/hip_bf16.h>

#define NA   40000
#define NS   20000
#define HID  256
#define NH   8
#define HD   32
#define NE   100000

typedef __attribute__((ext_vector_type(8)))  __bf16 v8bf;
typedef __attribute__((ext_vector_type(16))) __bf16 v16bf;
typedef __attribute__((ext_vector_type(8)))  float  v8f;

// ---------- helpers ----------
__device__ __forceinline__ __bf16 f2bf(float f) {
  unsigned u = __builtin_bit_cast(unsigned, f);
  u += 0x7FFFu + ((u >> 16) & 1u);              // round-to-nearest-even
  unsigned short h = (unsigned short)(u >> 16);
  return __builtin_bit_cast(__bf16, h);
}

__device__ __forceinline__ float gelu_exact(float x) {
  return 0.5f * x * (1.0f + erff(x * 0.70710678118654752f));
}

__device__ __forceinline__ unsigned fl_order(float f) {
  unsigned u = __builtin_bit_cast(unsigned, f);
  return (u & 0x80000000u) ? ~u : (u | 0x80000000u);
}
__device__ __forceinline__ float fl_unorder(unsigned m) {
  unsigned u = (m & 0x80000000u) ? (m ^ 0x80000000u) : ~m;
  return __builtin_bit_cast(float, u);
}

// ---------- weight pre-pack into WMMA B-fragment layout ----------
// Fragment (kt, nt): 32 lanes x 16 bf16, lane L holds col nt*16+(L&15),
// K = kt*32 + (L<16?0:16) + i.  Stored: [((kt*tilesN+nt)*32+lane)*16 + i].
__global__ void pack_b(const float* __restrict__ B, long ldb, long sB,
                       __bf16* __restrict__ out, long sOut,
                       int tilesK, int tilesN)
{
  int t = blockIdx.x * blockDim.x + threadIdx.x;
  int total = tilesK * tilesN * 32;
  if (t >= total) return;
  int lane = t & 31, tile = t >> 5;
  int kt = tile / tilesN, nt = tile % tilesN;
  B   += (long)blockIdx.y * sB;
  out += (long)blockIdx.y * sOut;
  int l16 = lane & 15, kb16 = (lane >> 4) * 16;
  const float* src = B + (long)(kt * 32 + kb16) * ldb + nt * 16 + l16;
  __bf16* dst = out + (long)t * 16;
#pragma unroll
  for (int i = 0; i < 16; ++i) dst[i] = f2bf(src[(long)i * ldb]);
}

// ---------- f32 -> bf16 elementwise (optionally fused exact GELU) ----------
__global__ void cvt_bf16(const float* __restrict__ in, __bf16* __restrict__ out,
                         long n, int doGelu)
{
  long i = (long)blockIdx.x * blockDim.x + threadIdx.x;
  if (i >= n) return;
  float v = in[i];
  if (doGelu) v = gelu_exact(v);
  out[i] = f2bf(v);
}

// ---------- dense WMMA GEMM with async-to-LDS B staging ----------
// Block = 8 waves = 256 rows x 64 cols of C.  All waves share one 64-col B strip:
// per K-step the 4KB strip is copied global->LDS with GLOBAL_LOAD_ASYNC_TO_LDS_B128
// (double-buffered, overlapped via ASYNCcnt), waves read fragments via ds_load_b128.
__global__ void gemm_wmma_lds(const __bf16* __restrict__ A, long lda,
                              const __bf16* __restrict__ Bp,
                              const float* __restrict__ bias,
                              float* __restrict__ Cf, __bf16* __restrict__ Cb,
                              long ldc, int tilesM2, int tilesN, int K)
{
  __shared__ __align__(32) __bf16 sB[2][4 * 512];   // 2 x 4KB double buffer

  int tid  = threadIdx.x;
  int wid  = tid >> 5;
  int lane = tid & 31;
  int tMw  = blockIdx.y * 8 + wid;
  bool active = (tMw < tilesM2);

  int l16 = lane & 15, hi = lane >> 4;
  int base8 = hi * 16;
  long arow0 = (long)(tMw * 32 + l16) * lda;
  long arow1 = arow0 + 16 * lda;

  unsigned ldsBase = (unsigned)(uintptr_t)&sB[0][0];
  long stripElems = ((long)blockIdx.x * 4) * 512;    // first fragment of this N strip

  auto issue = [&](int kt, int buf) {
    const __bf16* src = Bp + ((long)kt * tilesN * 512) + stripElems + (long)tid * 8;
    unsigned dst = ldsBase + (unsigned)buf * 4096u + (unsigned)tid * 16u;
    asm volatile("global_load_async_to_lds_b128 %0, %1, off"
                 :: "v"(dst), "v"((unsigned long long)(uintptr_t)src)
                 : "memory");
  };

  v8f acc[2][4];
#pragma unroll
  for (int s = 0; s < 2; ++s)
#pragma unroll
    for (int j = 0; j < 4; ++j) acc[s][j] = (v8f){};

  int nk = K / 32;
  issue(0, 0);
  for (int kt = 0; kt < nk; ++kt) {
    int buf = kt & 1;
    if (kt + 1 < nk) {
      issue(kt + 1, buf ^ 1);
      asm volatile("s_wait_asynccnt 0x1" ::: "memory");  // kt's copy done, next in flight
    } else {
      asm volatile("s_wait_asynccnt 0x0" ::: "memory");
    }
    __syncthreads();                                     // strip visible to all waves

    if (active) {
      const __bf16* Ap0 = A + arow0 + kt * 32 + base8;
      const __bf16* Ap1 = A + arow1 + kt * 32 + base8;
      v8bf a0l = *(const v8bf*)(Ap0);
      v8bf a0h = *(const v8bf*)(Ap0 + 8);
      v8bf a1l = *(const v8bf*)(Ap1);
      v8bf a1h = *(const v8bf*)(Ap1 + 8);
      v16bf a0 = __builtin_shufflevector(a0l, a0h, 0,1,2,3,4,5,6,7,8,9,10,11,12,13,14,15);
      v16bf a1 = __builtin_shufflevector(a1l, a1h, 0,1,2,3,4,5,6,7,8,9,10,11,12,13,14,15);
#pragma unroll
      for (int j = 0; j < 4; ++j) {
        v16bf bv = *(const v16bf*)&sB[buf][(j * 32 + lane) * 16];   // ds_load_b128 x2
        acc[0][j] = __builtin_amdgcn_wmma_f32_16x16x32_bf16(
            false, a0, false, bv, (short)0, acc[0][j], false, false);
        acc[1][j] = __builtin_amdgcn_wmma_f32_16x16x32_bf16(
            false, a1, false, bv, (short)0, acc[1][j], false, false);
      }
    }
    __syncthreads();                                     // done reading buf before overwrite
  }

  if (!active) return;
#pragma unroll
  for (int s = 0; s < 2; ++s) {
    int rowb = tMw * 32 + s * 16 + hi * 8;
#pragma unroll
    for (int j = 0; j < 4; ++j) {
      int col = blockIdx.x * 64 + j * 16 + l16;
      float bb = bias ? bias[col] : 0.0f;
#pragma unroll
      for (int r = 0; r < 8; ++r) {
        float v = acc[s][j][r] + bb;
        if (Cf) Cf[(long)(rowb + r) * ldc + col] = v;
        if (Cb) Cb[(long)(rowb + r) * ldc + col] = f2bf(v);
      }
    }
  }
}

// ---------- small-N WMMA GEMM (relation transforms, K=32, batched heads) ----------
__global__ void gemm_wmma2(const __bf16* __restrict__ A, long lda, long sA,
                           const __bf16* __restrict__ Bp, long sB,
                           float* __restrict__ Cf, long ldc, long sC,
                           int tilesM2, int tilesN, int K)
{
  int wid  = threadIdx.x >> 5;
  int lane = threadIdx.x & 31;
  int t = blockIdx.x * (blockDim.x >> 5) + wid;
  if (t >= tilesM2) return;                 // tilesN2 == 1 (N = 32)
  int tM = t;

  A  += (long)blockIdx.y * sA;
  Bp += (long)blockIdx.y * sB;
  Cf += (long)blockIdx.y * sC;

  int l16 = lane & 15, hi = lane >> 4;
  int base8 = hi * 16;
  long arow0 = (long)(tM * 32 + l16) * lda;
  long arow1 = arow0 + 16 * lda;

  v8f acc[2][2];
#pragma unroll
  for (int s = 0; s < 2; ++s)
#pragma unroll
    for (int j = 0; j < 2; ++j) acc[s][j] = (v8f){};

  int kt = 0;
  for (int k0 = 0; k0 < K; k0 += 32, ++kt) {
    const __bf16* Ap0 = A + arow0 + k0 + base8;
    const __bf16* Ap1 = A + arow1 + k0 + base8;
    v8bf a0l = *(const v8bf*)(Ap0);
    v8bf a0h = *(const v8bf*)(Ap0 + 8);
    v8bf a1l = *(const v8bf*)(Ap1);
    v8bf a1h = *(const v8bf*)(Ap1 + 8);
    v16bf a0 = __builtin_shufflevector(a0l, a0h, 0,1,2,3,4,5,6,7,8,9,10,11,12,13,14,15);
    v16bf a1 = __builtin_shufflevector(a1l, a1h, 0,1,2,3,4,5,6,7,8,9,10,11,12,13,14,15);
#pragma unroll
    for (int j = 0; j < 2; ++j) {
      const __bf16* bp = Bp + (((long)kt * tilesN + j) * 32 + lane) * 16;
      v16bf bv = *(const v16bf*)bp;
      acc[0][j] = __builtin_amdgcn_wmma_f32_16x16x32_bf16(
          false, a0, false, bv, (short)0, acc[0][j], false, false);
      acc[1][j] = __builtin_amdgcn_wmma_f32_16x16x32_bf16(
          false, a1, false, bv, (short)0, acc[1][j], false, false);
    }
  }

#pragma unroll
  for (int s = 0; s < 2; ++s) {
    int rowb = tM * 32 + s * 16 + hi * 8;
#pragma unroll
    for (int j = 0; j < 2; ++j) {
      int col = j * 16 + l16;
#pragma unroll
      for (int r = 0; r < 8; ++r)
        Cf[(long)(rowb + r) * ldc + col] = acc[s][j][r];
    }
  }
}

// ---------- edge kernels ----------
__global__ void zero_mem(float* __restrict__ p, long n) {
  long i = (long)blockIdx.x * blockDim.x + threadIdx.x;
  if (i < n) p[i] = 0.0f;
}

__global__ void edge_score(const float* __restrict__ qd, const float* __restrict__ kk,
                           const int* __restrict__ ei, const float* __restrict__ pr,
                           float* __restrict__ alpha, unsigned* __restrict__ mx, int nE)
{
  int t = blockIdx.x * blockDim.x + threadIdx.x;
  int e = t >> 3, h = t & 7;
  if (e >= nE) return;
  int s = ei[e], d = ei[nE + e];
  const float* qp = qd + (long)d * HID + h * HD;
  const float* kp = kk + (long)s * HID + h * HD;
  float acc = 0.f;
#pragma unroll
  for (int i = 0; i < HD; ++i) acc += qp[i] * kp[i];
  acc *= pr[h] * 0.17677669529663687f;       // 1/sqrt(32)
  alpha[(long)e * NH + h] = acc;
  atomicMax(&mx[(long)d * NH + h], fl_order(acc));
}

__global__ void edge_expsum(float* __restrict__ alpha, const int* __restrict__ ei,
                            const unsigned* __restrict__ mx, float* __restrict__ denom, int nE)
{
  int t = blockIdx.x * blockDim.x + threadIdx.x;
  int e = t >> 3, h = t & 7;
  if (e >= nE) return;
  int d = ei[nE + e];
  float ex = expf(alpha[(long)e * NH + h] - fl_unorder(mx[(long)d * NH + h]));
  alpha[(long)e * NH + h] = ex;
  atomicAdd(&denom[(long)d * NH + h], ex);
}

__global__ void edge_aggregate(const float* __restrict__ alpha, const float* __restrict__ denom,
                               const float* __restrict__ vv, const int* __restrict__ ei,
                               float* __restrict__ agg, int nE)
{
  long w = ((long)blockIdx.x * blockDim.x + threadIdx.x) >> 5;
  int lane = threadIdx.x & 31;
  long e = w >> 3; int h = (int)(w & 7);
  if (e >= nE) return;
  int s = ei[e], d = ei[nE + e];
  float attn = alpha[e * NH + h] / (denom[(long)d * NH + h] + 1e-16f);
  atomicAdd(&agg[(long)d * HID + h * HD + lane],
            attn * vv[(long)s * HID + h * HD + lane]);
}

__global__ void blend_relu_bf(const float* __restrict__ o, const float* __restrict__ skip,
                              float* __restrict__ xd, __bf16* __restrict__ xdb, long n)
{
  long i = (long)blockIdx.x * blockDim.x + threadIdx.x;
  if (i >= n) return;
  float sk = 1.0f / (1.0f + expf(-skip[0]));
  float v = sk * o[i] + (1.0f - sk) * xd[i];
  v = v > 0.0f ? v : 0.0f;
  xd[i] = v;
  xdb[i] = f2bf(v);
}

// ---------- host orchestration ----------
static inline void gemm_dense(hipStream_t st, const __bf16* A, long lda,
                              const __bf16* Bp, const float* bias,
                              float* Cf, __bf16* Cb, long ldc,
                              int M, int N, int K)
{
  int tilesM2 = M / 32, tilesN = N / 16, tilesN2 = tilesN / 4;
  dim3 grid(tilesN2, (tilesM2 + 7) / 8);
  gemm_wmma_lds<<<grid, 256, 0, st>>>(A, lda, Bp, bias, Cf, Cb, ldc, tilesM2, tilesN, K);
}

static inline void gemm_rel(hipStream_t st, const __bf16* A, long lda, long sA,
                            const __bf16* Bp, long sB,
                            float* Cf, long ldc, long sC, int M, int batch)
{
  int tilesM2 = M / 32;
  dim3 grid((tilesM2 + 7) / 8, batch);
  gemm_wmma2<<<grid, 256, 0, st>>>(A, lda, sA, Bp, sB, Cf, ldc, sC, tilesM2, 2, 32);
}

static inline void packW(hipStream_t st, const float* B, long ldb, long sB,
                         __bf16* out, long sOut, int K, int N, int batch)
{
  int threads = (K / 32) * (N / 16) * 32;
  dim3 grid((threads + 255) / 256, batch);
  pack_b<<<grid, 256, 0, st>>>(B, ldb, sB, out, sOut, K / 32, N / 16);
}

extern "C" void kernel_launch(void* const* d_in, const int* in_sizes, int n_in,
                              void* d_out, int out_size, void* d_ws, size_t ws_size,
                              hipStream_t stream)
{
  (void)in_sizes; (void)n_in; (void)out_size; (void)ws_size;

  const float* x_a = (const float*)d_in[0];
  const float* x_s = (const float*)d_in[1];
  const int* eiv[3] = {(const int*)d_in[2], (const int*)d_in[3], (const int*)d_in[4]};

  // params flattened in JAX sorted-dict order
  int pi = 5;
  auto P = [&]() { return (const float*)d_in[pi++]; };
  const float *embWa = P(), *embBa = P(), *embWs = P(), *embBs = P();
  struct LinT { const float *kW,*kb,*oW,*ob,*qW,*qb,*skip,*vW,*vb; };
  struct RelT { const float *a,*m,*pr; };
  LinT lin[2][2]; RelT rel[2][3];
  for (int l = 0; l < 2; ++l) {
    for (int t = 0; t < 2; ++t) {   // keys sorted: k,o,q,skip,v
      lin[l][t].kW = P(); lin[l][t].kb = P();
      lin[l][t].oW = P(); lin[l][t].ob = P();
      lin[l][t].qW = P(); lin[l][t].qb = P();
      lin[l][t].skip = P();
      lin[l][t].vW = P(); lin[l][t].vb = P();
    }
    for (int r = 0; r < 3; ++r) { rel[l][r].a = P(); rel[l][r].m = P(); rel[l][r].pr = P(); }
  }

  // -------- workspace carve (float units; sizes are multiples of 16) --------
  float* w = (float*)d_ws;
  auto alloc  = [&](long nf)  { float* r = w; w += nf; return r; };
  auto balloc = [&](long nbf) { return (__bf16*)alloc(nbf / 2); };

  float* xd_a  = alloc((long)NA * HID);
  float* xd_s  = alloc((long)NS * HID);
  float* q_a   = alloc((long)NA * HID);
  float* q_s   = alloc((long)NS * HID);
  float* kk_t  = alloc((long)NA * HID);                 // per-relation kk; later tmp_o
  float* vvb[3]= { alloc((long)NA * HID), alloc((long)NA * HID), alloc((long)NS * HID) };
  float* agg_a = alloc((long)NA * HID);
  float* agg_s = alloc((long)NS * HID);
  float* alpha[3] = { alloc((long)NE*NH), alloc((long)NE*NH), alloc((long)NE*NH) };
  unsigned* mx_a = (unsigned*)alloc((long)NA * NH);
  unsigned* mx_s = (unsigned*)alloc((long)NS * NH);
  float* den_a = alloc((long)NA * NH);
  float* den_s = alloc((long)NS * NH);
  float* tmp_o = kk_t;

  __bf16* x_a_bf  = balloc((long)NA * 128);
  __bf16* x_s_bf  = balloc((long)NS * 64);
  __bf16* xd_a_bf = balloc((long)NA * HID);
  __bf16* xd_s_bf = balloc((long)NS * HID);
  __bf16* k_a_bf  = balloc((long)NA * HID);
  __bf16* v_a_bf  = balloc((long)NA * HID);
  __bf16* k_s_bf  = balloc((long)NS * HID);
  __bf16* v_s_bf  = balloc((long)NS * HID);
  __bf16* agg_a_bf= balloc((long)NA * HID);
  __bf16* agg_s_bf= balloc((long)NS * HID);

  __bf16* pEmbWa = balloc(128L * 256);
  __bf16* pEmbWs = balloc(64L * 256);
  __bf16 *pLin[2][2][4];            // [layer][type][k,q,v,o]
  __bf16 *pRelA[2][3], *pRelM[2][3];
  for (int l = 0; l < 2; ++l) {
    for (int t = 0; t < 2; ++t)
      for (int j = 0; j < 4; ++j) pLin[l][t][j] = balloc(256L * 256);
    for (int r = 0; r < 3; ++r) { pRelA[l][r] = balloc(8L*32*32); pRelM[l][r] = balloc(8L*32*32); }
  }

  // -------- pack weights & convert inputs --------
  packW(stream, embWa, 256, 0, pEmbWa, 0, 128, 256, 1);
  packW(stream, embWs, 256, 0, pEmbWs, 0, 64, 256, 1);
  for (int l = 0; l < 2; ++l) {
    for (int t = 0; t < 2; ++t) {
      packW(stream, lin[l][t].kW, 256, 0, pLin[l][t][0], 0, 256, 256, 1);
      packW(stream, lin[l][t].qW, 256, 0, pLin[l][t][1], 0, 256, 256, 1);
      packW(stream, lin[l][t].vW, 256, 0, pLin[l][t][2], 0, 256, 256, 1);
      packW(stream, lin[l][t].oW, 256, 0, pLin[l][t][3], 0, 256, 256, 1);
    }
    for (int r = 0; r < 3; ++r) {
      packW(stream, rel[l][r].a, 32, 32*32, pRelA[l][r], 32*32, 32, 32, 8);
      packW(stream, rel[l][r].m, 32, 32*32, pRelM[l][r], 32*32, 32, 32, 8);
    }
  }
  cvt_bf16<<<(unsigned)(((long)NA*128 + 255)/256), 256, 0, stream>>>(x_a, x_a_bf, (long)NA*128, 0);
  cvt_bf16<<<(unsigned)(((long)NS*64  + 255)/256), 256, 0, stream>>>(x_s, x_s_bf, (long)NS*64, 0);

  // -------- embeddings --------
  gemm_dense(stream, x_a_bf, 128, pEmbWa, embBa, xd_a, xd_a_bf, 256, NA, 256, 128);
  gemm_dense(stream, x_s_bf,  64, pEmbWs, embBs, xd_s, xd_s_bf, 256, NS, 256,  64);

  const int rsrc[3] = {0, 0, 1};   // aa, as, sa
  const int rdst[3] = {0, 1, 0};

  for (int l = 0; l < 2; ++l) {
    auto Z = [&](float* p, long n) { zero_mem<<<(unsigned)((n + 255)/256), 256, 0, stream>>>(p, n); };
    Z((float*)mx_a, (long)NA*NH); Z((float*)mx_s, (long)NS*NH);
    Z(den_a, (long)NA*NH);        Z(den_s, (long)NS*NH);
    Z(agg_a, (long)NA*HID);       Z(agg_s, (long)NS*HID);

    // k/q/v projections: k,v -> bf16 only (feed relation GEMMs); q -> f32 (edge scores)
    gemm_dense(stream, xd_a_bf, 256, pLin[l][0][0], lin[l][0].kb, nullptr, k_a_bf, 256, NA, 256, 256);
    gemm_dense(stream, xd_a_bf, 256, pLin[l][0][1], lin[l][0].qb, q_a,     nullptr, 256, NA, 256, 256);
    gemm_dense(stream, xd_a_bf, 256, pLin[l][0][2], lin[l][0].vb, nullptr, v_a_bf, 256, NA, 256, 256);
    gemm_dense(stream, xd_s_bf, 256, pLin[l][1][0], lin[l][1].kb, nullptr, k_s_bf, 256, NS, 256, 256);
    gemm_dense(stream, xd_s_bf, 256, pLin[l][1][1], lin[l][1].qb, q_s,     nullptr, 256, NS, 256, 256);
    gemm_dense(stream, xd_s_bf, 256, pLin[l][1][2], lin[l][1].vb, nullptr, v_s_bf, 256, NS, 256, 256);

    long ntEH = (long)NE * NH;
    for (int r = 0; r < 3; ++r) {
      const __bf16* ks = rsrc[r] == 0 ? k_a_bf : k_s_bf;
      const __bf16* vs = rsrc[r] == 0 ? v_a_bf : v_s_bf;
      int Nsrc         = rsrc[r] == 0 ? NA : NS;
      const float* qd  = rdst[r] == 0 ? q_a : q_s;
      unsigned* mxd    = rdst[r] == 0 ? mx_a : mx_s;
      gemm_rel(stream, ks, 256, 32, pRelA[l][r], 32*32, kk_t,   256, 32, Nsrc, 8);
      gemm_rel(stream, vs, 256, 32, pRelM[l][r], 32*32, vvb[r], 256, 32, Nsrc, 8);
      edge_score<<<(unsigned)((ntEH + 255)/256), 256, 0, stream>>>(
          qd, kk_t, eiv[r], rel[l][r].pr, alpha[r], mxd, NE);
    }
    for (int r = 0; r < 3; ++r) {
      unsigned* mxd = rdst[r] == 0 ? mx_a : mx_s;
      float* dend   = rdst[r] == 0 ? den_a : den_s;
      edge_expsum<<<(unsigned)((ntEH + 255)/256), 256, 0, stream>>>(alpha[r], eiv[r], mxd, dend, NE);
    }
    for (int r = 0; r < 3; ++r) {
      const float* dend = rdst[r] == 0 ? den_a : den_s;
      float* aggd       = rdst[r] == 0 ? agg_a : agg_s;
      long nt = ntEH * 32;
      edge_aggregate<<<(unsigned)((nt + 255)/256), 256, 0, stream>>>(
          alpha[r], dend, vvb[r], eiv[r], aggd, NE);
    }

    // GELU pass -> bf16, output projection, skip-blend + relu
    cvt_bf16<<<(unsigned)(((long)NA*HID + 255)/256), 256, 0, stream>>>(agg_a, agg_a_bf, (long)NA*HID, 1);
    cvt_bf16<<<(unsigned)(((long)NS*HID + 255)/256), 256, 0, stream>>>(agg_s, agg_s_bf, (long)NS*HID, 1);
    gemm_dense(stream, agg_a_bf, 256, pLin[l][0][3], lin[l][0].ob, tmp_o, nullptr, 256, NA, 256, 256);
    blend_relu_bf<<<(unsigned)(((long)NA*HID + 255)/256), 256, 0, stream>>>(
        tmp_o, lin[l][0].skip, xd_a, xd_a_bf, (long)NA*HID);
    gemm_dense(stream, agg_s_bf, 256, pLin[l][1][3], lin[l][1].ob, tmp_o, nullptr, 256, NS, 256, 256);
    blend_relu_bf<<<(unsigned)(((long)NS*HID + 255)/256), 256, 0, stream>>>(
        tmp_o, lin[l][1].skip, xd_s, xd_s_bf, (long)NS*HID);
  }

  hipMemcpyAsync(d_out, xd_a, (size_t)NA * HID * sizeof(float),
                 hipMemcpyDeviceToDevice, stream);
  hipMemcpyAsync((float*)d_out + (long)NA * HID, xd_s, (size_t)NS * HID * sizeof(float),
                 hipMemcpyDeviceToDevice, stream);
}
